// BAKTTime_87840671138348
// MI455X (gfx1250) — matrix operand order repaired and checked by reading, not compiled, not run on hardware
//
#include <hip/hip_runtime.h>
#include <hip/hip_bf16.h>

// Problem constants
#define B_    64
#define S_    512
#define D_    512
#define H_    8
#define DK_   64
#define NROW  (B_ * S_)   // 32768

typedef __attribute__((ext_vector_type(16))) _Float16 v16h;
typedef __attribute__((ext_vector_type(8)))  _Float16 h8;
typedef __attribute__((ext_vector_type(8)))  float    v8f;

__device__ __forceinline__ v8f wmma16(v16h a, v16h b, v8f c) {
  // D = A(16x32,f16) * B(32x16,f16) + C(16x16,f32)
  return __builtin_amdgcn_wmma_f32_16x16x32_f16(false, a, false, b, (short)0, c,
                                                false, false);
}

__device__ __forceinline__ v8f v8f_zero() {
  v8f z;
#pragma unroll
  for (int i = 0; i < 8; ++i) z[i] = 0.0f;
  return z;
}

__device__ __forceinline__ h8 h8_zero() {
  h8 z;
#pragma unroll
  for (int i = 0; i < 8; ++i) z[i] = (_Float16)0.0f;
  return z;
}

__device__ __forceinline__ v16h combine_a(h8 lo, h8 hi) {
  v16h a;
#pragma unroll
  for (int i = 0; i < 8; ++i) { a[i] = lo[i]; a[i + 8] = hi[i]; }
  return a;
}

// A-fragment (16x32 f16): lane l -> row m0+(l&15); g=l>>4; element e:
//   e<8  -> K = k0 + g*8 + e
//   e>=8 -> K = k0 + 16 + g*8 + (e-8)
__device__ __forceinline__ v16h load_a_frag(const _Float16* __restrict__ base,
                                            int lda, int m0, int k0, int lane) {
  int m = m0 + (lane & 15);
  int g = lane >> 4;
  const _Float16* row = base + (size_t)m * lda;
  h8 lo = *(const h8*)(row + k0 + g * 8);
  h8 hi = *(const h8*)(row + k0 + 16 + g * 8);
  return combine_a(lo, hi);
}

// B-fragment (32x16 f16) from row-major W where B[k][n] = W[n][k]:
// lane l -> col n0+(l&15); g=l>>4; element e:
//   e<8 -> K = k0 + g*8 + e; e>=8 -> K = k0 + 16 + g*8 + (e-8)
__device__ __forceinline__ v16h load_b_frag(const _Float16* __restrict__ base,
                                            int ldb, int n0, int k0, int lane) {
  int n = n0 + (lane & 15);
  int g = lane >> 4;
  const _Float16* col = base + (size_t)n * ldb;
  h8 lo = *(const h8*)(col + k0 + g * 8);
  h8 hi = *(const h8*)(col + k0 + 16 + g * 8);
  return combine_a(lo, hi);
}

__device__ __forceinline__ float half16_max(float x) {
#pragma unroll
  for (int m = 1; m < 16; m <<= 1) x = fmaxf(x, __shfl_xor(x, m, 32));
  return x;
}
__device__ __forceinline__ float half16_sum(float x) {
#pragma unroll
  for (int m = 1; m < 16; m <<= 1) x += __shfl_xor(x, m, 32);
  return x;
}

// ---------------------------------------------------------------- conversions
__global__ __launch_bounds__(256) void cvt_f16_kernel(const float* __restrict__ in,
                                                      _Float16* __restrict__ out,
                                                      int n) {
  int i = blockIdx.x * 256 + threadIdx.x;
  if (i < n) out[i] = (_Float16)in[i];
}

// conv_w [n][c][k] (f32) -> convWh [k][n][c] (f16)
__global__ __launch_bounds__(256) void convw_kernel(const float* __restrict__ cw,
                                                    _Float16* __restrict__ out) {
  int i = blockIdx.x * 256 + threadIdx.x;
  if (i < 3 * D_ * D_) {
    int k = i / (D_ * D_);
    int rem = i - k * (D_ * D_);
    int n = rem >> 9;
    int c = rem & (D_ - 1);
    out[i] = (_Float16)cw[((size_t)n * D_ + c) * 3 + k];
  }
}

// ------------------------------------------- fused conv + freq-mix + layernorm
// One block: 16 rows (same batch) x 512 cols. 8 waves, 4 n-tiles each.
__global__ __launch_bounds__(256) void freq_ln_kernel(
    const _Float16* __restrict__ xh, const float* __restrict__ x,
    const _Float16* __restrict__ convWh, const float* __restrict__ conv_b,
    const float* __restrict__ sqrt_beta, const float* __restrict__ ln_w,
    const float* __restrict__ ln_b, _Float16* __restrict__ h_out) {
  __shared__ __align__(32) float lds_t[16][D_];
  __shared__ float row_mean[16], row_rstd[16];

  int tid = threadIdx.x, lane = tid & 31, wave = tid >> 5, g = lane >> 4;
  int M0 = blockIdx.x * 16;
  int b = M0 >> 9, s0 = M0 & (S_ - 1);

  v8f acc[4];
#pragma unroll
  for (int t = 0; t < 4; ++t) acc[t] = v8f_zero();

  for (int k = 0; k < 3; ++k) {
    const _Float16* Wk = convWh + (size_t)k * D_ * D_;
    int sl = s0 + (lane & 15) + k - 2;  // shifted seq index (may be <0)
    const _Float16* arow = xh + ((size_t)b * S_ + sl) * D_;
    for (int kc = 0; kc < D_; kc += 32) {
      h8 lo = h8_zero(), hi = h8_zero();
      if (sl >= 0) {
        lo = *(const h8*)(arow + kc + g * 8);
        hi = *(const h8*)(arow + kc + 16 + g * 8);
      }
      v16h a = combine_a(lo, hi);
#pragma unroll
      for (int t = 0; t < 4; ++t) {
        v16h bf = load_b_frag(Wk, D_, wave * 64 + t * 16, kc, lane);
        acc[t] = wmma16(a, bf, acc[t]);
      }
    }
  }

  // epilogue: low -> emb + residual into LDS
#pragma unroll
  for (int t = 0; t < 4; ++t) {
    int n = wave * 64 + t * 16 + (lane & 15);
    float cb = conv_b[n];
    float sb = sqrt_beta[n];
    float sb2 = sb * sb;
#pragma unroll
    for (int r = 0; r < 8; ++r) {
      int row = r + 8 * g;  // local row in tile
      float low = acc[t][r] + cb;
      float xv = x[((size_t)b * S_ + s0 + row) * D_ + n];
      lds_t[row][n] = low + sb2 * (xv - low) + xv;  // emb + x
    }
  }
  __syncthreads();

  // layernorm: 16 threads per row (contiguous lane groups within a wave)
  {
    int row = tid >> 4, sub = tid & 15;
    float sum = 0.f, sumsq = 0.f;
    for (int c = sub; c < D_; c += 16) {
      float v = lds_t[row][c];
      sum += v;
      sumsq += v * v;
    }
#pragma unroll
    for (int m = 1; m < 16; m <<= 1) {
      sum += __shfl_xor(sum, m, 32);
      sumsq += __shfl_xor(sumsq, m, 32);
    }
    if (sub == 0) {
      float mean = sum * (1.0f / D_);
      float var = sumsq * (1.0f / D_) - mean * mean;
      row_mean[row] = mean;
      row_rstd[row] = rsqrtf(fmaxf(var, 0.0f) + 1e-12f);
    }
  }
  __syncthreads();

  for (int idx = tid; idx < 16 * D_; idx += 256) {
    int rr = idx >> 9, c = idx & (D_ - 1);
    float v = (lds_t[rr][c] - row_mean[rr]) * row_rstd[rr] * ln_w[c] + ln_b[c];
    h_out[((size_t)b * S_ + s0 + rr) * D_ + c] = (_Float16)v;
  }
}

// ------------------------------------------------------------ QKV projections
// grid: (2048 m-tiles, 2 n-groups of 256, 3 matrices); block 128 (4 waves x 4 tiles)
__global__ __launch_bounds__(128) void qkv_kernel(
    const _Float16* __restrict__ h, const _Float16* __restrict__ Wqkvh,
    const float* __restrict__ bq, const float* __restrict__ bk,
    const float* __restrict__ bv, _Float16* __restrict__ qh,
    _Float16* __restrict__ kh, _Float16* __restrict__ vT) {
  int tid = threadIdx.x, lane = tid & 31, wave = tid >> 5, g = lane >> 4;
  int which = blockIdx.z;
  int M0 = blockIdx.x * 16;
  int nbase = blockIdx.y * 256 + wave * 64;
  const _Float16* W = Wqkvh + (size_t)which * D_ * D_;
  const float* bias = (which == 0) ? bq : (which == 1) ? bk : bv;

  v8f acc[4];
#pragma unroll
  for (int t = 0; t < 4; ++t) acc[t] = v8f_zero();

  for (int kc = 0; kc < D_; kc += 32) {
    v16h a = load_a_frag(h, D_, M0, kc, lane);
#pragma unroll
    for (int t = 0; t < 4; ++t) {
      v16h bf = load_b_frag(W, D_, nbase + t * 16, kc, lane);
      acc[t] = wmma16(a, bf, acc[t]);
    }
  }

  int b = M0 >> 9, s0 = M0 & (S_ - 1);
#pragma unroll
  for (int t = 0; t < 4; ++t) {
    int n = nbase + t * 16 + (lane & 15);
    float bias_v = bias[n];
    int hh = n >> 6, dk = n & (DK_ - 1);
    size_t bh = (size_t)b * H_ + hh;
#pragma unroll
    for (int r = 0; r < 8; ++r) {
      int s = s0 + r + 8 * g;
      _Float16 y = (_Float16)(acc[t][r] + bias_v);
      if (which == 0)      qh[(bh * S_ + s) * DK_ + dk] = y;
      else if (which == 1) kh[(bh * S_ + s) * DK_ + dk] = y;
      else                 vT[(bh * DK_ + dk) * S_ + s] = y;  // transposed
    }
  }
}

// ------------------------------------------------- causal attention (flash)
// grid: (S/128, B*H); block 256 = 8 waves; wave handles 16 q-rows.
__global__ __launch_bounds__(256) void attn_kernel(
    const _Float16* __restrict__ qh, const _Float16* __restrict__ kh,
    const _Float16* __restrict__ vT, _Float16* __restrict__ ctxh) {
  __shared__ __align__(32) _Float16 lds_p[8][16][32];

  int tid = threadIdx.x, lane = tid & 31, wave = tid >> 5, g = lane >> 4;
  int bh = blockIdx.y;
  int q0 = blockIdx.x * 128 + wave * 16;

  const _Float16* qbase = qh + (size_t)bh * S_ * DK_;
  const _Float16* kbase = kh + (size_t)bh * S_ * DK_;
  const _Float16* vbase = vT + (size_t)bh * DK_ * S_;

  v16h qa0 = load_a_frag(qbase, DK_, q0, 0, lane);
  v16h qa1 = load_a_frag(qbase, DK_, q0, 32, lane);

  v8f acc[4];
#pragma unroll
  for (int t = 0; t < 4; ++t) acc[t] = v8f_zero();
  float mrow[8], lrow[8];
#pragma unroll
  for (int r = 0; r < 8; ++r) { mrow[r] = -1e30f; lrow[r] = 0.0f; }

  int nchunk = q0 / 32 + 1;  // causal: kv chunks of 32 up to q-tile
  for (int j = 0; j < nchunk; ++j) {
    int kv0 = j * 32;
    float s0v[8], s1v[8];
    // scores subtile 0 (kv cols kv0..kv0+15)
    {
      v8f sa = v8f_zero();
      sa = wmma16(qa0, load_b_frag(kbase, DK_, kv0, 0, lane), sa);
      sa = wmma16(qa1, load_b_frag(kbase, DK_, kv0, 32, lane), sa);
      int kvc = kv0 + (lane & 15);
#pragma unroll
      for (int r = 0; r < 8; ++r) {
        int qr = q0 + r + 8 * g;
        float sv = sa[r] * 0.125f;          // 1/sqrt(64)
        s0v[r] = (kvc > qr) ? -1e30f : sv;  // causal mask
      }
    }
    // scores subtile 1 (kv cols kv0+16..kv0+31)
    {
      v8f sa = v8f_zero();
      sa = wmma16(qa0, load_b_frag(kbase, DK_, kv0 + 16, 0, lane), sa);
      sa = wmma16(qa1, load_b_frag(kbase, DK_, kv0 + 16, 32, lane), sa);
      int kvc = kv0 + 16 + (lane & 15);
#pragma unroll
      for (int r = 0; r < 8; ++r) {
        int qr = q0 + r + 8 * g;
        float sv = sa[r] * 0.125f;
        s1v[r] = (kvc > qr) ? -1e30f : sv;
      }
    }
    // online softmax update + stage P into LDS (C-layout -> A-layout)
#pragma unroll
    for (int r = 0; r < 8; ++r) {
      float rm = half16_max(fmaxf(s0v[r], s1v[r]));
      float mn = fmaxf(mrow[r], rm);
      float sc = __expf(mrow[r] - mn);
      float p0 = __expf(s0v[r] - mn);
      float p1 = __expf(s1v[r] - mn);
      lrow[r] = lrow[r] * sc + half16_sum(p0 + p1);
      mrow[r] = mn;
      acc[0][r] *= sc; acc[1][r] *= sc; acc[2][r] *= sc; acc[3][r] *= sc;
      lds_p[wave][r + 8 * g][lane & 15]        = (_Float16)p0;
      lds_p[wave][r + 8 * g][16 + (lane & 15)] = (_Float16)p1;
    }
    // wave-local LDS visibility (LDS ops in-order per wave; wait for safety)
    asm volatile("s_wait_dscnt 0" ::: "memory");
    h8 plo = *(const h8*)(&lds_p[wave][lane & 15][g * 8]);
    h8 phi = *(const h8*)(&lds_p[wave][lane & 15][16 + g * 8]);
    asm volatile("s_wait_dscnt 0" ::: "memory");
    v16h pA = combine_a(plo, phi);
    // P(16x32) @ V(32x64): 4 d-tiles from transposed V (contiguous loads)
#pragma unroll
    for (int t = 0; t < 4; ++t) {
      v16h vb = load_b_frag(vbase, S_, t * 16, kv0, lane);
      acc[t] = wmma16(pA, vb, acc[t]);
    }
  }

  // epilogue: normalize, zero_pad first query row, write ctx (f16)
  int b = bh >> 3, hh = bh & (H_ - 1);
#pragma unroll
  for (int r = 0; r < 8; ++r) {
    int qr = q0 + r + 8 * g;
    float inv = 1.0f / lrow[r];
    if (qr == 0) inv = 0.0f;  // attn[:, :, 0, :] = 0
#pragma unroll
    for (int t = 0; t < 4; ++t) {
      float cv = acc[t][r] * inv;
      ctxh[((size_t)b * S_ + qr) * D_ + hh * DK_ + t * 16 + (lane & 15)] =
          (_Float16)cv;
    }
  }
}

// ------------------------------------------------------------ output proj
__global__ __launch_bounds__(128) void oproj_kernel(
    const _Float16* __restrict__ ctx, const _Float16* __restrict__ Woh,
    const float* __restrict__ bo, float* __restrict__ out) {
  int tid = threadIdx.x, lane = tid & 31, wave = tid >> 5, g = lane >> 4;
  int M0 = blockIdx.x * 16;
  int nbase = blockIdx.y * 256 + wave * 64;

  v8f acc[4];
#pragma unroll
  for (int t = 0; t < 4; ++t) acc[t] = v8f_zero();

  for (int kc = 0; kc < D_; kc += 32) {
    v16h a = load_a_frag(ctx, D_, M0, kc, lane);
#pragma unroll
    for (int t = 0; t < 4; ++t) {
      v16h bf = load_b_frag(Woh, D_, nbase + t * 16, kc, lane);
      acc[t] = wmma16(a, bf, acc[t]);
    }
  }
#pragma unroll
  for (int t = 0; t < 4; ++t) {
    int n = nbase + t * 16 + (lane & 15);
    float bias_v = bo[n];
#pragma unroll
    for (int r = 0; r < 8; ++r) {
      int m = M0 + r + 8 * g;
      out[(size_t)m * D_ + n] = acc[t][r] + bias_v;
    }
  }
}

// ---------------------------------------------------------------------------
extern "C" void kernel_launch(void* const* d_in, const int* in_sizes, int n_in,
                              void* d_out, int out_size, void* d_ws,
                              size_t ws_size, hipStream_t stream) {
  const float* x         = (const float*)d_in[0];
  const float* conv_w    = (const float*)d_in[1];
  const float* conv_b    = (const float*)d_in[2];
  const float* sqrt_beta = (const float*)d_in[3];
  const float* ln_w      = (const float*)d_in[4];
  const float* ln_b      = (const float*)d_in[5];
  const float* Wq        = (const float*)d_in[6];
  const float* bq        = (const float*)d_in[7];
  const float* Wk        = (const float*)d_in[8];
  const float* bk        = (const float*)d_in[9];
  const float* Wv        = (const float*)d_in[10];
  const float* bv        = (const float*)d_in[11];
  const float* Wo        = (const float*)d_in[12];
  const float* bo        = (const float*)d_in[13];
  float* out = (float*)d_out;

  char* ws = (char*)d_ws;
  const size_t BIG = (size_t)NROW * D_ * sizeof(_Float16);  // 33,554,432 B
  _Float16* xh     = (_Float16*)(ws + 0 * BIG);
  _Float16* ctxh   = xh;  // alias: xh is dead after freq_ln, attn writes ctx here
  _Float16* hbuf   = (_Float16*)(ws + 1 * BIG);
  _Float16* qh     = (_Float16*)(ws + 2 * BIG);
  _Float16* kh     = (_Float16*)(ws + 3 * BIG);
  _Float16* vT     = (_Float16*)(ws + 4 * BIG);
  _Float16* Wqkvh  = (_Float16*)(ws + 5 * BIG);
  _Float16* Woh    = Wqkvh + 3 * D_ * D_;
  _Float16* convWh = Woh + (size_t)D_ * D_;

  // 1) f32 -> f16 conversions / weight re-layout
  {
    int nx = NROW * D_;
    cvt_f16_kernel<<<(nx + 255) / 256, 256, 0, stream>>>(x, xh, nx);
    int nw = D_ * D_;
    cvt_f16_kernel<<<(nw + 255) / 256, 256, 0, stream>>>(Wq, Wqkvh, nw);
    cvt_f16_kernel<<<(nw + 255) / 256, 256, 0, stream>>>(Wk, Wqkvh + nw, nw);
    cvt_f16_kernel<<<(nw + 255) / 256, 256, 0, stream>>>(Wv, Wqkvh + 2 * nw, nw);
    cvt_f16_kernel<<<(nw + 255) / 256, 256, 0, stream>>>(Wo, Woh, nw);
    int nc = 3 * D_ * D_;
    convw_kernel<<<(nc + 255) / 256, 256, 0, stream>>>(conv_w, convWh);
  }

  // 2) fused causal-conv + frequency mix + layernorm -> h (f16)
  freq_ln_kernel<<<NROW / 16, 256, 0, stream>>>(xh, x, convWh, conv_b,
                                                sqrt_beta, ln_w, ln_b, hbuf);

  // 3) QKV projections
  qkv_kernel<<<dim3(NROW / 16, 2, 3), 128, 0, stream>>>(hbuf, Wqkvh, bq, bk,
                                                        bv, qh, kh, vT);

  // 4) causal attention with zero_pad
  attn_kernel<<<dim3(S_ / 128, B_ * H_), 256, 0, stream>>>(qh, kh, vT, ctxh);

  // 5) output projection (f32 out)
  oproj_kernel<<<dim3(NROW / 16, 2), 128, 0, stream>>>(ctxh, Woh, bo, out);

  (void)in_sizes; (void)n_in; (void)out_size; (void)ws_size;
}
